// PointNetEncoderV2_7816840479273
// MI455X (gfx1250) — matrix operand order repaired and checked by reading, not compiled, not run on hardware
//
#include <hip/hip_runtime.h>

typedef __attribute__((ext_vector_type(16))) _Float16 v16h;
typedef __attribute__((ext_vector_type(8)))  _Float16 v8h;
typedef __attribute__((ext_vector_type(8)))  float    v8f;
typedef __attribute__((ext_vector_type(4)))  float    v4f;

#define N_TOTAL 65536
#define BC      76                 // B*C = 4*19
#define SUBT    128                // points per subtile
#define NSUB    (N_TOTAL / SUBT)   // 512 subtiles
#define ZSZ     (BC * 64)          // 4864 running-max entries
#define NEGBIG  (-1e30f)
#define WSTR    72                 // f16 stride for weights / activations (144B, 16B-aligned, bank-spread)
#define FSTR    68                 // f32 stride for features (272B, 16B-aligned)

// Flip to 0 if the async-to-LDS inline asm is rejected by the assembler.
#define USE_ASYNC_LDS 1

// ---- LDS layout (bytes) for the fused encoder kernel ----
#define OFF_WGT2 0                 // 64x72 f16 = 9216
#define OFF_WGT3 9216              // 64x72 f16 = 9216
#define OFF_H1   18432             // 128 cols x 72 f16 = 18432 (column-major)
#define OFF_H2   36864             // 18432
#define OFF_FEAT 55296             // 128 x 68 f32 = 34816 (n-major)
#define OFF_MASK 90112             // 76x128 i32 = 38912
#define OFF_ZACC 129024            // 76x64 f32 = 19456
#define OFF_W1   148480            // 128 f32 = 512
#define OFF_B1   148992            // 256
#define OFF_B2   149248            // 256
#define OFF_B3   149504            // 256
#define SMEM_BYTES 149760

__device__ __forceinline__ float lrelu(float v) { return v >= 0.f ? v : 0.01f * v; }

// One 64x128 layer: D = W(64x64,f16) @ H(64x128,f16) + bias, leaky-ReLU.
// Weights row-major (stride WSTR), activations column-major (stride WSTR):
// every WMMA fragment is two contiguous 16B runs -> ds_load_b128 pairs.
// Wave `wave` owns output columns [wave*16, wave*16+16).
template <bool F16OUT>
__device__ __forceinline__ void mlp_layer(const _Float16* __restrict__ wgt,
                                          const _Float16* __restrict__ hin,
                                          const float*    __restrict__ bias,
                                          _Float16*       __restrict__ hout,
                                          float*          __restrict__ fout,
                                          int lane, int wave)
{
    const int lm   = lane & 15;
    const int half = lane >> 4;
    const int bcol = wave * 16 + lm;

#pragma unroll
    for (int m = 0; m < 4; ++m) {
        v8f acc = {};
        const int arow = m * 16 + lm;
#pragma unroll
        for (int kc = 0; kc < 2; ++kc) {
            // A fragment: a[i] = W[arow][kc*32 + half*8 + i], a[8+i] = +16  (ISA 16-bit A 16x32)
            const _Float16* ap = wgt + arow * WSTR + kc * 32 + half * 8;
            v8h alo = *(const v8h*)(ap);
            v8h ahi = *(const v8h*)(ap + 16);
            v16h a = __builtin_shufflevector(alo, ahi, 0, 1, 2, 3, 4, 5, 6, 7,
                                             8, 9, 10, 11, 12, 13, 14, 15);
            // B fragment: b[i] = H[K = kc*32 + half*8 + i][bcol], b[8+i] = +16  (same k order as A)
            const _Float16* bp = hin + bcol * WSTR + kc * 32 + half * 8;
            v8h blo = *(const v8h*)(bp);
            v8h bhi = *(const v8h*)(bp + 16);
            v16h b = __builtin_shufflevector(blo, bhi, 0, 1, 2, 3, 4, 5, 6, 7,
                                             8, 9, 10, 11, 12, 13, 14, 15);
            acc = __builtin_amdgcn_wmma_f32_16x16x32_f16(
                false, a, false, b, (short)0, acc, false, false);
        }
        // Epilogue. C/D layout: VGPR j -> row m*16 + j + 8*half at column bcol.
        if (F16OUT) {
            v8h hv;
#pragma unroll
            for (int j = 0; j < 8; ++j) {
                const int row = m * 16 + 8 * half + j;
                hv[j] = (_Float16)lrelu(acc[j] + bias[row]);
            }
            *(v8h*)(hout + bcol * WSTR + m * 16 + 8 * half) = hv;
        } else {
            v4f f0, f1;
#pragma unroll
            for (int j = 0; j < 4; ++j) {
                const int row = m * 16 + 8 * half + j;
                f0[j] = lrelu(acc[j] + bias[row]);
                f1[j] = lrelu(acc[j + 4] + bias[row + 4]);
            }
            float* fp = fout + bcol * FSTR + m * 16 + 8 * half;
            *(v4f*)(fp)     = f0;
            *(v4f*)(fp + 4) = f1;
        }
    }
}

// Fused: per 128-point subtile -> async mask copy overlapped with
// layer1 (VALU) -> layer2/3 (WMMA) -> masked running max into LDS zacc[76][64].
__global__ void __launch_bounds__(256)
encode_max_kernel(const int* __restrict__ x, const float* __restrict__ grid,
                  const float* __restrict__ w1, const float* __restrict__ b1,
                  const float* __restrict__ w2, const float* __restrict__ b2,
                  const float* __restrict__ w3, const float* __restrict__ b3,
                  float* __restrict__ part, int nblocks)
{
    extern __shared__ char smem[];
    _Float16* wgt2 = (_Float16*)(smem + OFF_WGT2);
    _Float16* wgt3 = (_Float16*)(smem + OFF_WGT3);
    _Float16* h1   = (_Float16*)(smem + OFF_H1);
    _Float16* h2   = (_Float16*)(smem + OFF_H2);
    float*    feat = (float*)(smem + OFF_FEAT);
    int*      msk  = (int*)(smem + OFF_MASK);
    float*    zacc = (float*)(smem + OFF_ZACC);
    float*    w1s  = (float*)(smem + OFF_W1);
    float*    b1s  = (float*)(smem + OFF_B1);
    float*    b2s  = (float*)(smem + OFF_B2);
    float*    b3s  = (float*)(smem + OFF_B3);

    const int t    = threadIdx.x;
    const int lane = t & 31;
    const int wave = t >> 5;

    // Stage weights (f32 -> f16, padded stride) + biases; init running max.
    for (int i = t; i < 4096; i += 256) {
        const int r = i >> 6, k = i & 63;
        wgt2[r * WSTR + k] = (_Float16)w2[i];
        wgt3[r * WSTR + k] = (_Float16)w3[i];
    }
    for (int i = t; i < 128; i += 256) w1s[i] = w1[i];
    if (t < 64) { b1s[t] = b1[t]; b2s[t] = b2[t]; b3s[t] = b3[t]; }
    for (int i = t; i < ZSZ; i += 256) zacc[i] = NEGBIG;
    __syncthreads();

    for (int s = blockIdx.x; s < NSUB; s += nblocks) {
        const int nbase = s * SUBT;

        // ---- Stage mask tile (76 rows x 128 i32) into LDS ----
#if USE_ASYNC_LDS
        // Async DMA-style copy: 2432 x 16B chunks; overlaps with the MLP below.
        for (int i = t; i < BC * SUBT / 4; i += 256) {
            const int row = i >> 5, c4 = i & 31;            // 32 x b128 per row
            const int* gp = x + (size_t)row * N_TOTAL + nbase + c4 * 4;
            const unsigned lds_lo = (unsigned)(size_t)(const void*)(msk + i * 4);
            asm volatile("global_load_async_to_lds_b128 %0, %1, off"
                         :: "v"(lds_lo), "v"(gp) : "memory");
            if (s + nblocks < NSUB)
                __builtin_prefetch(gp + nblocks * SUBT, 0, 1);
        }
#else
        for (int i = t; i < BC * SUBT; i += 256) {
            const int bc = i >> 7, n = i & 127;
            msk[i] = x[(size_t)bc * N_TOTAL + nbase + n];
            if (s + nblocks < NSUB)
                __builtin_prefetch(&x[(size_t)bc * N_TOTAL + nbase + nblocks * SUBT + n], 0, 1);
        }
#endif

        // ---- Layer 1: [2 -> 64] pointwise, store f16 column-major ----
        for (int i = t; i < 64 * SUBT; i += 256) {
            const int n = i >> 6, k = i & 63;
            const float gx = grid[nbase + n];
            const float gy = grid[N_TOTAL + nbase + n];
            h1[n * WSTR + k] =
                (_Float16)lrelu(w1s[2 * k] * gx + w1s[2 * k + 1] * gy + b1s[k]);
        }
        __syncthreads();

        // ---- Layer 2 (WMMA) -> h2 (f16 col-major), Layer 3 (WMMA) -> feat ----
        mlp_layer<true >(wgt2, h1, b2s, h2, nullptr, lane, wave);
        __syncthreads();
        mlp_layer<false>(wgt3, h2, b3s, nullptr, feat, lane, wave);

#if USE_ASYNC_LDS
        asm volatile("s_wait_asynccnt 0x0" ::: "memory");
#endif
        __syncthreads();

        // ---- Masked running max: thread owns (k = t&63, bc stride 4) ----
        {
            const int k = t & 63, bcq = t >> 6;
            for (int bc = bcq; bc < BC; bc += 4) {
                float m = zacc[bc * 64 + k];
                const int* mr = msk + bc * SUBT;
#pragma unroll 4
                for (int n = 0; n < SUBT; ++n) {
                    if (mr[n]) m = fmaxf(m, feat[n * FSTR + k]);
                }
                zacc[bc * 64 + k] = m;
            }
        }
        __syncthreads();
    }

    for (int i = t; i < ZSZ; i += 256) part[(size_t)blockIdx.x * ZSZ + i] = zacc[i];
}

// Reduce per-block partials; substitute zero_feat (zero point through the MLP,
// i.e. biases through the 3 layers) for empty masks.
__global__ void __launch_bounds__(256)
reduce_kernel(const float* __restrict__ part, int nblocks,
              const float* __restrict__ b1,
              const float* __restrict__ w2, const float* __restrict__ b2,
              const float* __restrict__ w3, const float* __restrict__ b3,
              float* __restrict__ zfinal)
{
    __shared__ float zf1[64], zf2[64], zf[64];
    const int t = threadIdx.x;
    if (t < 64) zf1[t] = lrelu(b1[t]);
    __syncthreads();
    if (t < 64) {
        float s = b2[t];
        for (int j = 0; j < 64; ++j) s += w2[t * 64 + j] * zf1[j];
        zf2[t] = lrelu(s);
    }
    __syncthreads();
    if (t < 64) {
        float s = b3[t];
        for (int j = 0; j < 64; ++j) s += w3[t * 64 + j] * zf2[j];
        zf[t] = lrelu(s);
    }
    __syncthreads();

    for (int idx = t; idx < ZSZ; idx += 256) {
        float m = NEGBIG;
        for (int blk = 0; blk < nblocks; ++blk)
            m = fmaxf(m, part[(size_t)blk * ZSZ + idx]);
        if (m < -1e29f) m = zf[idx & 63];   // empty mask -> zero_feat
        zfinal[idx] = m;
    }
}

// EqualLinear: out[b,o] = fcb[o] + (1/sqrt(1216)) * dot(z[b,:], fcw[o,:]); repeat 18x.
__global__ void __launch_bounds__(256)
fc_kernel(const float* __restrict__ zfinal, const float* __restrict__ fcw,
          const float* __restrict__ fcb, float* __restrict__ out)
{
    const int gid = blockIdx.x * 256 + threadIdx.x;
    if (gid >= 4 * 512) return;
    const int b = gid >> 9, o = gid & 511;
    const float* fl = zfinal + b * 1216;
    const float* wr = fcw + (size_t)o * 1216;
    float s = 0.f;
    for (int j = 0; j < 1216; ++j) s += fl[j] * wr[j];
    s = s * (1.0f / 34.871191f) + fcb[o];   // 1/sqrt(19*64)
#pragma unroll
    for (int r = 0; r < 18; ++r) out[(b * 18 + r) * 512 + o] = s;
}

extern "C" void kernel_launch(void* const* d_in, const int* in_sizes, int n_in,
                              void* d_out, int out_size, void* d_ws, size_t ws_size,
                              hipStream_t stream)
{
    const int*   x    = (const int*)d_in[0];
    const float* grid = (const float*)d_in[1];
    const float* w1   = (const float*)d_in[2];
    const float* b1   = (const float*)d_in[3];
    const float* w2   = (const float*)d_in[4];
    const float* b2   = (const float*)d_in[5];
    const float* w3   = (const float*)d_in[6];
    const float* b3   = (const float*)d_in[7];
    const float* fcw  = (const float*)d_in[8];
    const float* fcb  = (const float*)d_in[9];
    float* out = (float*)d_out;
    float* ws  = (float*)d_ws;

    // Deterministic block count: fit per-block partials + zfinal in workspace.
    const size_t per = (size_t)ZSZ * sizeof(float);
    int nblocks = 256;
    if (ws_size < (size_t)(nblocks + 1) * per) {
        nblocks = (int)(ws_size / per) - 1;
        if (nblocks < 1) nblocks = 1;
    }
    float* part   = ws;
    float* zfinal = ws + (size_t)nblocks * ZSZ;

    (void)hipFuncSetAttribute(reinterpret_cast<const void*>(encode_max_kernel),
                              hipFuncAttributeMaxDynamicSharedMemorySize, SMEM_BYTES);

    encode_max_kernel<<<nblocks, 256, SMEM_BYTES, stream>>>(
        x, grid, w1, b1, w2, b2, w3, b3, part, nblocks);
    reduce_kernel<<<1, 256, 0, stream>>>(part, nblocks, b1, w2, b2, w3, b3, zfinal);
    fc_kernel<<<8, 256, 0, stream>>>(zfinal, fcw, fcb, out);
}
